// RNN_34093450396226
// MI455X (gfx1250) — matrix operand order repaired and checked by reading, not compiled, not run on hardware
//
#include <hip/hip_runtime.h>
#include <hip/hip_bf16.h>
#include <math.h>

// CDNA5 (gfx1250) fused 3-layer tanh-RNN.
// One wave32 owns a 16-row batch tile for all T=512 steps and all 3 layers.
// Recurrent/ih matmuls: v_wmma_f32_16x16x32_f16, H=20 padded to 32 (N and K).
// Weights live in per-lane v16h B-fragments for the whole kernel.
// Activation uses the hardware v_tanh_f32 (TRANS32, co-executes with WMMA)
// instead of branchy libm tanhf.

typedef __attribute__((ext_vector_type(16))) _Float16 v16h;
typedef __attribute__((ext_vector_type(8)))  _Float16 v8h;
typedef __attribute__((ext_vector_type(8)))  float    v8f;

#define BATCH 8192
#define TLEN  512
#define HDIM  20
#define HPAD  32
#define WAVES_PER_BLOCK 8

union AU { v16h v; v8h h[2]; _Float16 e[16]; };

__device__ __forceinline__ v8f wmma_f16(v16h a, v16h b, v8f c) {
  // (neg_a, A, neg_b, B, c_mod, C, reuse_a, reuse_b)
  return __builtin_amdgcn_wmma_f32_16x16x32_f16(false, a, false, b,
                                                (short)0, c, false, false);
}

__device__ __forceinline__ void ds_fence() {
  asm volatile("s_wait_dscnt 0" ::: "memory");
}

// Branch-free tanh. Prefer the CDNA5 hardware instruction (v_tanh_f32);
// fall back to exp2/rcp rational form (all TRANS/VALU, no EXEC branching).
__device__ __forceinline__ float tanh_fast(float x) {
#if __has_builtin(__builtin_amdgcn_tanhf)
  return __builtin_amdgcn_tanhf(x);
#else
  const float ax = __builtin_fabsf(x);
  // tanh(|x|) = (1 - e) / (1 + e),  e = 2^(-2*log2(e)*|x|)
  const float e = __builtin_amdgcn_exp2f(-2.8853900817779268f * ax);
#if __has_builtin(__builtin_amdgcn_rcpf)
  const float r = (1.0f - e) * __builtin_amdgcn_rcpf(1.0f + e);
#else
  const float r = (1.0f - e) / (1.0f + e);
#endif
  return __builtin_copysignf(r, x);
#endif
}

// Build B-matrix (32x16 f16) fragment for tile tn of W (HDIM x HDIM row-major),
// used as h @ W^T, i.e. B[k][n] = W[n][k].
// Assumed layout: lane L holds column N = L%16; lanes 0-15 hold K=0..15,
// lanes 16-31 hold K=16..31, K ascending through the 16 half elements.
// Pad rows/columns are exactly zero -> padding stays zero through the whole
// recurrence (so no per-element masking is needed anywhere).
__device__ __forceinline__ v16h load_B_tile(const float* __restrict__ W,
                                            int lane, int tn) {
  AU u;
  const int n = (lane & 15) + 16 * tn;
  const int kbase = (lane < 16) ? 0 : 16;
#pragma unroll
  for (int e = 0; e < 16; ++e) {
    const int k = kbase + e;
    float w = (n < HDIM && k < HDIM) ? W[n * HDIM + k] : 0.0f;
    u.e[e] = (_Float16)w;
  }
  return u.v;
}

// Load A-matrix (16x32 f16) fragment from LDS staging tile (16 rows x HPAD
// halves, row-major). Contiguous k order, same as load_B_tile: lane L (<16) row M=L holds K=0..15;
// lane L+16 row M=L holds K=16..31. Two ds_load_b128 per lane.
__device__ __forceinline__ v16h load_A_lds(const _Float16* __restrict__ stag,
                                           int lane) {
  AU u;
  const int m  = lane & 15;
  const int k0 = (lane < 16) ? 0 : 16;
  u.h[0] = *(const v8h*)(stag + m * HPAD + k0);
  u.h[1] = *(const v8h*)(stag + m * HPAD + k0 + 8);
  return u.v;
}

// Apply tanh to a D fragment (f32, M in vgpr idx, N in lane) and store as f16
// rows into the LDS staging tile. Pad columns carry acc==0 -> tanh(0)==0.
__device__ __forceinline__ void store_h_lds(_Float16* __restrict__ stag,
                                            int lane, int tn, v8f acc) {
  const int n    = (lane & 15) + 16 * tn;
  const int mofs = (lane < 16) ? 0 : 8;
#pragma unroll
  for (int i = 0; i < 8; ++i) {
    stag[(i + mofs) * HPAD + n] = (_Float16)tanh_fast(acc[i]);
  }
}

__global__ __launch_bounds__(32 * WAVES_PER_BLOCK)
void rnn3_fused_wmma(const float* __restrict__ x,
                     const float* __restrict__ w_ih0, const float* __restrict__ w_hh0,
                     const float* __restrict__ b_ih0, const float* __restrict__ b_hh0,
                     const float* __restrict__ w_ih1, const float* __restrict__ w_hh1,
                     const float* __restrict__ b_ih1, const float* __restrict__ b_hh1,
                     const float* __restrict__ w_ih2, const float* __restrict__ w_hh2,
                     const float* __restrict__ b_ih2, const float* __restrict__ b_hh2,
                     const float* __restrict__ fc_w, const float* __restrict__ fc_b,
                     float* __restrict__ out) {
  __shared__ _Float16 stag_all[WAVES_PER_BLOCK][16 * HPAD];

  const int lane  = threadIdx.x & 31;
  const int wave  = threadIdx.x >> 5;
  const int tile  = blockIdx.x * WAVES_PER_BLOCK + wave;
  const int bbase = tile * 16;
  _Float16* stag  = stag_all[wave];

  // ---- stage weight B-fragments into registers (reused for all 512 steps) ----
  const v16h Bhh0_0 = load_B_tile(w_hh0, lane, 0), Bhh0_1 = load_B_tile(w_hh0, lane, 1);
  const v16h Bih1_0 = load_B_tile(w_ih1, lane, 0), Bih1_1 = load_B_tile(w_ih1, lane, 1);
  const v16h Bhh1_0 = load_B_tile(w_hh1, lane, 0), Bhh1_1 = load_B_tile(w_hh1, lane, 1);
  const v16h Bih2_0 = load_B_tile(w_ih2, lane, 0), Bih2_1 = load_B_tile(w_ih2, lane, 1);
  const v16h Bhh2_0 = load_B_tile(w_hh2, lane, 0), Bhh2_1 = load_B_tile(w_hh2, lane, 1);

  // per-lane layer-0 input projection weight and per-layer bias sums (per N-tile)
  const int n0 = lane & 15, n1 = (lane & 15) + 16;
  const float wi0_0 = (n0 < HDIM) ? w_ih0[n0] : 0.0f;
  const float wi0_1 = (n1 < HDIM) ? w_ih0[n1] : 0.0f;
  const float bs0_0 = (n0 < HDIM) ? (b_ih0[n0] + b_hh0[n0]) : 0.0f;
  const float bs0_1 = (n1 < HDIM) ? (b_ih0[n1] + b_hh0[n1]) : 0.0f;
  const float bs1_0 = (n0 < HDIM) ? (b_ih1[n0] + b_hh1[n0]) : 0.0f;
  const float bs1_1 = (n1 < HDIM) ? (b_ih1[n1] + b_hh1[n1]) : 0.0f;
  const float bs2_0 = (n0 < HDIM) ? (b_ih2[n0] + b_hh2[n0]) : 0.0f;
  const float bs2_1 = (n1 < HDIM) ? (b_ih2[n1] + b_hh2[n1]) : 0.0f;

  // hidden-state A-fragments, h(-1) = 0
  AU aH0, aH1, aH2;
#pragma unroll
  for (int e = 0; e < 16; ++e) {
    aH0.e[e] = (_Float16)0; aH1.e[e] = (_Float16)0; aH2.e[e] = (_Float16)0;
  }

  const float* xrow = x + (size_t)(bbase + (lane & 15)) * TLEN;  // x is (B,T,1)
  const int srcbase = (lane < 16) ? 0 : 8;

#pragma unroll 1
  for (int t = 0; t < TLEN; ++t) {
    // x value for batch row (lane&15); broadcast to M-per-vgpr layout via shfl
    const float xv = xrow[t];
    if ((t & 63) == 0 && t + 64 < TLEN) __builtin_prefetch(xrow + t + 64, 0, 0);
    float xm[8];
#pragma unroll
    for (int i = 0; i < 8; ++i) xm[i] = __shfl(xv, srcbase + i, 32);

    v8f acc0, acc1;

    // ---------------- layer 0: h0 = tanh(x*w_ih0 + b + h0 @ w_hh0^T) --------
#pragma unroll
    for (int i = 0; i < 8; ++i) {
      acc0[i] = xm[i] * wi0_0 + bs0_0;
      acc1[i] = xm[i] * wi0_1 + bs0_1;
    }
    acc0 = wmma_f16(aH0.v, Bhh0_0, acc0);
    acc1 = wmma_f16(aH0.v, Bhh0_1, acc1);
    ds_fence();                                   // WAR vs previous A loads
    store_h_lds(stag, lane, 0, acc0);
    store_h_lds(stag, lane, 1, acc1);
    ds_fence();
    aH0.v = load_A_lds(stag, lane);               // h0(t) in A layout

    // ---------------- layer 1: h1 = tanh(b + h0 @ w_ih1^T + h1 @ w_hh1^T) ---
#pragma unroll
    for (int i = 0; i < 8; ++i) { acc0[i] = bs1_0; acc1[i] = bs1_1; }
    acc0 = wmma_f16(aH0.v, Bih1_0, acc0);
    acc1 = wmma_f16(aH0.v, Bih1_1, acc1);
    acc0 = wmma_f16(aH1.v, Bhh1_0, acc0);
    acc1 = wmma_f16(aH1.v, Bhh1_1, acc1);
    ds_fence();
    store_h_lds(stag, lane, 0, acc0);
    store_h_lds(stag, lane, 1, acc1);
    ds_fence();
    aH1.v = load_A_lds(stag, lane);               // h1(t)

    // ---------------- layer 2: h2 = tanh(b + h1 @ w_ih2^T + h2 @ w_hh2^T) ---
#pragma unroll
    for (int i = 0; i < 8; ++i) { acc0[i] = bs2_0; acc1[i] = bs2_1; }
    acc0 = wmma_f16(aH1.v, Bih2_0, acc0);
    acc1 = wmma_f16(aH1.v, Bih2_1, acc1);
    acc0 = wmma_f16(aH2.v, Bhh2_0, acc0);
    acc1 = wmma_f16(aH2.v, Bhh2_1, acc1);
    ds_fence();
    store_h_lds(stag, lane, 0, acc0);
    store_h_lds(stag, lane, 1, acc1);
    ds_fence();
    aH2.v = load_A_lds(stag, lane);               // h2(t)
  }

  // ---- FC head: out[b] = h2(T-1) . fc_w + fc_b (staging holds h2(T-1)) ----
  ds_fence();
  if (lane < 16) {
    float s = fc_b[0];
#pragma unroll
    for (int k = 0; k < HDIM; ++k)
      s += (float)stag[lane * HPAD + k] * fc_w[k];
    out[bbase + lane] = s;
  }
}

extern "C" void kernel_launch(void* const* d_in, const int* in_sizes, int n_in,
                              void* d_out, int out_size, void* d_ws, size_t ws_size,
                              hipStream_t stream) {
  const float* x     = (const float*)d_in[0];
  const float* w_ih0 = (const float*)d_in[1];
  const float* w_hh0 = (const float*)d_in[2];
  const float* b_ih0 = (const float*)d_in[3];
  const float* b_hh0 = (const float*)d_in[4];
  const float* w_ih1 = (const float*)d_in[5];
  const float* w_hh1 = (const float*)d_in[6];
  const float* b_ih1 = (const float*)d_in[7];
  const float* b_hh1 = (const float*)d_in[8];
  const float* w_ih2 = (const float*)d_in[9];
  const float* w_hh2 = (const float*)d_in[10];
  const float* b_ih2 = (const float*)d_in[11];
  const float* b_hh2 = (const float*)d_in[12];
  const float* fc_w  = (const float*)d_in[13];
  const float* fc_b  = (const float*)d_in[14];
  float* out = (float*)d_out;

  const int tiles  = BATCH / 16;                 // 512 wave tiles
  const int blocks = tiles / WAVES_PER_BLOCK;    // 64 blocks of 256 threads
  hipLaunchKernelGGL(rnn3_fused_wmma, dim3(blocks), dim3(32 * WAVES_PER_BLOCK),
                     0, stream,
                     x, w_ih0, w_hh0, b_ih0, b_hh0,
                     w_ih1, w_hh1, b_ih1, b_hh1,
                     w_ih2, w_hh2, b_ih2, b_hh2,
                     fc_w, fc_b, out);
}
